// CustomAttention_80169859547219
// MI455X (gfx1250) — matrix operand order repaired and checked by reading, not compiled, not run on hardware
//
#include <hip/hip_runtime.h>
#include <stdint.h>

typedef __bf16 v16bf __attribute__((ext_vector_type(16)));
typedef float  v8f   __attribute__((ext_vector_type(8)));

using u16 = unsigned short;
using u32 = unsigned int;

__device__ __forceinline__ u16 f32_to_bf16(float f) {
  u32 x = __float_as_uint(f);
  x += 0x7FFFu + ((x >> 16) & 1u);   // round-to-nearest-even
  return (u16)(x >> 16);
}

// ---------------------------------------------------------------------------
// fp32 -> bf16 conversion (grid-stride, float4 loads, 8B packed stores)
// ---------------------------------------------------------------------------
__global__ __launch_bounds__(256) void cvt_f32_bf16(const float* __restrict__ s,
                                                    u16* __restrict__ d, long n) {
  long stride = (long)gridDim.x * blockDim.x * 4;
  for (long i = ((long)blockIdx.x * blockDim.x + threadIdx.x) * 4; i < n; i += stride) {
    float4 f = *(const float4*)(s + i);
    union { u16 r[4]; uint2 p; } t;
    t.r[0] = f32_to_bf16(f.x); t.r[1] = f32_to_bf16(f.y);
    t.r[2] = f32_to_bf16(f.z); t.r[3] = f32_to_bf16(f.w);
    *(uint2*)(d + i) = t.p;
  }
}

// ---------------------------------------------------------------------------
// Generic bf16 WMMA GEMM:  C[M,N] = A[M,K] * op(B) (+ bias)
//   B_IS_KXN=false : B stored [N,K] row-major (A*B^T), staged directly
//   B_IS_KXN=true  : B stored [K,N] row-major (A*B),  transposed while staging
// Tile: 128x64x32, 256 threads (8 wave32), each wave = 2x2 grid of 16x16x32
// v_wmma_f32_16x16x32_bf16 fragments.
// ---------------------------------------------------------------------------
constexpr int BM = 128, BN = 64, BK = 32;
constexpr int LDK = BK + 4;   // 36 elements = 72B padded row (conflict-free)

template<bool B_IS_KXN, bool OUT_BF16, bool HAS_BIAS>
__global__ __launch_bounds__(256) void gemm_bf16(
    const u16* __restrict__ A, const u16* __restrict__ Bm,
    const float* __restrict__ bias, void* __restrict__ Cp,
    int M, int N, int K, int lda, int ldb, int ldc,
    long sA, long sB, long sC)
{
  __shared__ u16 As[BM * LDK];
  __shared__ u16 Bs[BN * LDK];

  const int t    = threadIdx.x;
  const int wid  = t >> 5;
  const int lane = t & 31;
  const int half = lane >> 4;   // wave32: lane halves per ISA fragment layout
  const int ml   = lane & 15;

  const int m0 = blockIdx.x * BM;
  const int n0 = blockIdx.y * BN;
  const int bz = blockIdx.z;

  const u16* Ab = A  + (size_t)bz * sA;
  const u16* Bb = Bm + (size_t)bz * sB;

  const int wm = (wid >> 1) * 32;   // 4 waves along M
  const int wn = (wid & 1)  * 32;   // 2 waves along N

  v8f acc[2][2];
  acc[0][0] = v8f{}; acc[0][1] = v8f{}; acc[1][0] = v8f{}; acc[1][1] = v8f{};

  // staging coordinates
  const int ar = t >> 1;            // 0..127 : A row
  const int ac = (t & 1) * 16;      // 0/16   : A k-chunk
  const int br = t >> 2;            // 0..63  : B row   (NT path)
  const int bc = (t & 3) * 8;       // 0..24  : B k-chunk
  const int bkr = t >> 3;           // 0..31  : B k-row (NN path)
  const int bnc = (t & 7) * 8;      // 0..56  : B n-chunk

  for (int kk = 0; kk < K; kk += BK) {
    // ---- stage A tile: BM x BK bf16 ----
    {
      const u16* src = Ab + (size_t)(m0 + ar) * lda + kk + ac;
      u16* dst = &As[ar * LDK + ac];
      uint2 d0 = *(const uint2*)(src + 0);
      uint2 d1 = *(const uint2*)(src + 4);
      uint2 d2 = *(const uint2*)(src + 8);
      uint2 d3 = *(const uint2*)(src + 12);
      *(uint2*)(dst + 0)  = d0;
      *(uint2*)(dst + 4)  = d1;
      *(uint2*)(dst + 8)  = d2;
      *(uint2*)(dst + 12) = d3;
      if (kk + BK < K)   // emits global_prefetch_b8 for next K tile
        __builtin_prefetch(src + BK, 0, 1);
    }
    // ---- stage B tile as [n][k] ----
    if (!B_IS_KXN) {
      const u16* src = Bb + (size_t)(n0 + br) * ldb + kk + bc;
      u16* dst = &Bs[br * LDK + bc];
      uint2 d0 = *(const uint2*)(src + 0);
      uint2 d1 = *(const uint2*)(src + 4);
      *(uint2*)(dst + 0) = d0;
      *(uint2*)(dst + 4) = d1;
    } else {
      const u16* src = Bb + (size_t)(kk + bkr) * ldb + n0 + bnc;
      union { uint4 q; u16 s[8]; } tmp;
      tmp.q = *(const uint4*)src;
#pragma unroll
      for (int e = 0; e < 8; ++e)
        Bs[(bnc + e) * LDK + bkr] = tmp.s[e];
    }
    __syncthreads();

    // ---- fragment loads (ISA 7.12.2 16-bit A 16x32 / B 32x16 layouts) ----
    union Frag { v16bf v; uint2 u2[4]; };
    Frag a[2], b[2];
#pragma unroll
    for (int i = 0; i < 2; ++i) {
      const u16* p = &As[(wm + 16 * i + ml) * LDK + half * 8];
      a[i].u2[0] = *(const uint2*)(p + 0);    // k = half*8 .. +3
      a[i].u2[1] = *(const uint2*)(p + 4);    // k = half*8+4 .. +7
      a[i].u2[2] = *(const uint2*)(p + 16);   // k = 16+half*8 ..
      a[i].u2[3] = *(const uint2*)(p + 20);
    }
#pragma unroll
    for (int j = 0; j < 2; ++j) {
      const u16* p = &Bs[(wn + 16 * j + ml) * LDK + half * 8];
      b[j].u2[0] = *(const uint2*)(p + 0);    // k = half*8 .. +7, then 16+half*8 .. +7
      b[j].u2[1] = *(const uint2*)(p + 4);
      b[j].u2[2] = *(const uint2*)(p + 16);
      b[j].u2[3] = *(const uint2*)(p + 20);
    }
#pragma unroll
    for (int i = 0; i < 2; ++i)
#pragma unroll
      for (int j = 0; j < 2; ++j)
        acc[i][j] = __builtin_amdgcn_wmma_f32_16x16x32_bf16(
            false, a[i].v, false, b[j].v, (short)0, acc[i][j], false, false);
    __syncthreads();
  }

  // ---- store C (f32 16x16 layout: row = 8*half + r, col = lane&15) ----
#pragma unroll
  for (int i = 0; i < 2; ++i) {
#pragma unroll
    for (int j = 0; j < 2; ++j) {
      const int col = n0 + wn + 16 * j + ml;
      float bv = HAS_BIAS ? bias[col] : 0.0f;
#pragma unroll
      for (int r = 0; r < 8; ++r) {
        const int row = m0 + wm + 16 * i + 8 * half + r;
        float v = acc[i][j][r] + bv;
        if (OUT_BF16)
          ((u16*)Cp)[(size_t)bz * sC + (size_t)row * ldc + col] = f32_to_bf16(v);
        else
          ((float*)Cp)[(size_t)bz * sC + (size_t)row * ldc + col] = v;
      }
    }
  }
}

// ---------------------------------------------------------------------------
// Row softmax over 2048 f32 scores -> bf16 weights. One block per row.
// ---------------------------------------------------------------------------
__global__ __launch_bounds__(256) void softmax_rows(const float* __restrict__ S,
                                                    u16* __restrict__ W, int n) {
  __shared__ float red[8];
  const long row = blockIdx.x;
  const float* src = S + row * (long)n;
  u16* dst = W + row * (long)n;
  const int t = threadIdx.x;
  const int base = t * 8;

  float4 v0 = *(const float4*)(src + base);
  float4 v1 = *(const float4*)(src + base + 4);
  float x[8] = {v0.x, v0.y, v0.z, v0.w, v1.x, v1.y, v1.z, v1.w};

  float m = x[0];
#pragma unroll
  for (int i = 1; i < 8; ++i) m = fmaxf(m, x[i]);
  for (int o = 16; o > 0; o >>= 1) m = fmaxf(m, __shfl_xor(m, o, 32));
  if ((t & 31) == 0) red[t >> 5] = m;
  __syncthreads();
  float gm = red[0];
#pragma unroll
  for (int i = 1; i < 8; ++i) gm = fmaxf(gm, red[i]);
  __syncthreads();

  float s = 0.f;
#pragma unroll
  for (int i = 0; i < 8; ++i) { x[i] = __expf(x[i] - gm); s += x[i]; }
  for (int o = 16; o > 0; o >>= 1) s += __shfl_xor(s, o, 32);
  if ((t & 31) == 0) red[t >> 5] = s;
  __syncthreads();
  float gs = 0.f;
#pragma unroll
  for (int i = 0; i < 8; ++i) gs += red[i];
  float inv = 1.0f / gs;

  union { u16 r[8]; uint4 q; } o;
#pragma unroll
  for (int i = 0; i < 8; ++i) o.r[i] = f32_to_bf16(x[i] * inv);
  *(uint4*)(dst + base) = o.q;
}

// ---------------------------------------------------------------------------
extern "C" void kernel_launch(void* const* d_in, const int* in_sizes, int n_in,
                              void* d_out, int out_size, void* d_ws, size_t ws_size,
                              hipStream_t stream) {
  (void)in_sizes; (void)n_in; (void)out_size; (void)ws_size;
  const int Bb = 8, SQ = 2048, SKV = 2048, D = 1024;

  const float* target = (const float*)d_in[0];
  const float* nontgt = (const float*)d_in[1];
  const float* Wq = (const float*)d_in[2];
  const float* bq = (const float*)d_in[3];
  const float* Wk = (const float*)d_in[4];
  const float* bk = (const float*)d_in[5];
  const float* Wv = (const float*)d_in[6];
  const float* bv = (const float*)d_in[7];
  float* out = (float*)d_out;

  // workspace carve-up (all sizes naturally 256B aligned)
  char* w = (char*)d_ws;
  auto alloc = [&](size_t bytes) { void* p = (void*)w; w += (bytes + 255) & ~(size_t)255; return p; };
  const size_t nTok = (size_t)Bb * SQ * D;      // 16,777,216
  const size_t nSc  = (size_t)Bb * SQ * SKV;    // 33,554,432
  u16* tgt16 = (u16*)alloc(nTok * 2);
  u16* ntg16 = (u16*)alloc(nTok * 2);
  u16* wq16  = (u16*)alloc((size_t)D * D * 2);
  u16* wk16  = (u16*)alloc((size_t)D * D * 2);
  u16* wv16  = (u16*)alloc((size_t)D * D * 2);
  u16* qb    = (u16*)alloc(nTok * 2);
  u16* kb    = (u16*)alloc(nTok * 2);
  u16* vb    = (u16*)alloc(nTok * 2);
  float* sc  = (float*)alloc(nSc * 4);
  u16* wt    = (u16*)alloc(nSc * 2);

  // 1) fp32 -> bf16 conversions
  cvt_f32_bf16<<<4096, 256, 0, stream>>>(target, tgt16, (long)nTok);
  cvt_f32_bf16<<<4096, 256, 0, stream>>>(nontgt, ntg16, (long)nTok);
  cvt_f32_bf16<<<1024, 256, 0, stream>>>(Wq, wq16, (long)D * D);
  cvt_f32_bf16<<<1024, 256, 0, stream>>>(Wk, wk16, (long)D * D);
  cvt_f32_bf16<<<1024, 256, 0, stream>>>(Wv, wv16, (long)D * D);

  dim3 blk(256);
  // 2) projections: Y = X * W^T + b  (bf16 out)
  gemm_bf16<false, true, true><<<dim3(128, 16, 1), blk, 0, stream>>>(
      tgt16, wq16, bq, qb, Bb * SQ, D, D, D, D, D, 0, 0, 0);
  gemm_bf16<false, true, true><<<dim3(128, 16, 1), blk, 0, stream>>>(
      ntg16, wk16, bk, kb, Bb * SKV, D, D, D, D, D, 0, 0, 0);
  gemm_bf16<false, true, true><<<dim3(128, 16, 1), blk, 0, stream>>>(
      ntg16, wv16, bv, vb, Bb * SKV, D, D, D, D, D, 0, 0, 0);

  // 3) scores = q * k^T  (f32 out, batched)
  gemm_bf16<false, false, false><<<dim3(16, 32, 8), blk, 0, stream>>>(
      qb, kb, nullptr, sc, SQ, SKV, D, D, D, SKV,
      (long)SQ * D, (long)SKV * D, (long)SQ * SKV);

  // 4) softmax rows -> bf16 weights
  softmax_rows<<<Bb * SQ, 256, 0, stream>>>(sc, wt, SKV);

  // 5) out = weights * V  (B stored [K,N] -> transpose-stage path)
  gemm_bf16<true, false, false><<<dim3(16, 16, 8), blk, 0, stream>>>(
      wt, vb, nullptr, out, SQ, D, SKV, SKV, D, D,
      (long)SQ * SKV, (long)SKV * D, (long)SQ * D);
}
